// MIDMs_75969381532260
// MI455X (gfx1250) — matrix operand order repaired and checked by reading, not compiled, not run on hardware
//
#include <hip/hip_runtime.h>

typedef __attribute__((ext_vector_type(16))) _Float16 v16h;
typedef __attribute__((ext_vector_type(8)))  _Float16 v8h;
typedef __attribute__((ext_vector_type(8)))  float    v8f;
typedef __attribute__((ext_vector_type(4)))  unsigned int u32x4;
typedef __attribute__((ext_vector_type(8)))  int      i32x8;
typedef __attribute__((ext_vector_type(4)))  int      i32x4;

#define B_ 2
#define C_ 256
#define L_ 4096

#if defined(__gfx1250__) && __has_builtin(__builtin_amdgcn_tensor_load_to_lds)
#define USE_TDM 1
#else
#define USE_TDM 0
#endif

// ---------------------------------------------------------------------------
// Per-position channel normalization: x[B,C,H,W] (f32) -> y[B,L,C] (f16).
// ---------------------------------------------------------------------------
__global__ __launch_bounds__(256) void norm_kernel(const float* __restrict__ x,
                                                   _Float16* __restrict__ y) {
  const int wid  = (blockIdx.x * blockDim.x + threadIdx.x) >> 5;  // wave index
  const int lane = threadIdx.x & 31;
  const int b    = wid / L_;
  const int pos  = wid % L_;
  const int c0   = lane * 8;

  const size_t base = (size_t)b * C_ * L_ + pos;
  float v[8];
  float s = 0.f, s2 = 0.f;
#pragma unroll
  for (int i = 0; i < 8; ++i) {
    v[i] = x[base + (size_t)(c0 + i) * L_];
    s  += v[i];
    s2 += v[i] * v[i];
  }
#pragma unroll
  for (int m = 1; m < 32; m <<= 1) {
    s  += __shfl_xor(s,  m, 32);
    s2 += __shfl_xor(s2, m, 32);
  }
  const float mean = s * (1.0f / C_);
  const float inv  = rsqrtf(fmaxf(s2 - s * s * (1.0f / C_), 1e-20f));

  v8h o;
#pragma unroll
  for (int i = 0; i < 8; ++i) o[i] = (_Float16)((v[i] - mean) * inv);
  *(v8h*)&y[((size_t)b * L_ + pos) * C_ + c0] = o;
}

// ---------------------------------------------------------------------------
// Pack V matrices channel-major f16 with channel padding to a multiple of 16.
// ---------------------------------------------------------------------------
__global__ __launch_bounds__(256) void pack_src_v(const float* __restrict__ f,
                                                  _Float16* __restrict__ V) {
  const int idx = blockIdx.x * blockDim.x + threadIdx.x;  // over B*16*L
  const int pos = idx % L_;
  const int ch  = (idx / L_) % 16;
  const int b   = idx / (16 * L_);
  float val = (ch < 4) ? f[((size_t)b * 4 + ch) * L_ + pos] : 0.f;
  V[idx] = (_Float16)val;
}

__global__ __launch_bounds__(256) void pack_ref_v(const float* __restrict__ rf,
                                                  const float* __restrict__ rs,
                                                  _Float16* __restrict__ V) {
  const int idx = blockIdx.x * blockDim.x + threadIdx.x;  // over B*160*L
  const int pos = idx % L_;
  const int ch  = (idx / L_) % 160;
  const int b   = idx / (160 * L_);
  float val = 0.f;
  if (ch < 4)        val = rf[((size_t)b * 4 + ch) * L_ + pos];
  else if (ch < 155) val = rs[((size_t)b * 151 + (ch - 4)) * L_ + pos];
  V[idx] = (_Float16)val;
}

#if USE_TDM
// ---------------------------------------------------------------------------
// TDM descriptor helpers (D# group layout per CDNA5 ISA ch.8).
// data_size = 3 (8-byte units) for both: all tiles are multiples of 8 bytes.
// amdgpu-toolchain builtin form (6 args): (g0, g1, g2, g3, extra_i32x8, cpol).
// ---------------------------------------------------------------------------
__device__ __forceinline__ void tdm_load_1d(const void* gptr, void* lptr,
                                            unsigned tile_units,
                                            unsigned tensor_units) {
  const unsigned long long ga = (unsigned long long)(uintptr_t)gptr;
  const unsigned la = (unsigned)(uintptr_t)lptr;
  u32x4 g0;
  g0[0] = 1u;                                               // count=1, user D#
  g0[1] = la;                                               // lds_addr
  g0[2] = (unsigned)ga;                                     // global_addr lo
  g0[3] = (unsigned)((ga >> 32) & 0x01FFFFFFu) | (2u << 30);// addr hi | type=2
  i32x8 g1;
  g1[0] = (int)(3u << 16);                                  // data_size=8B
  g1[1] = (int)((tensor_units & 0xFFFFu) << 16);            // tensor_dim0 lo16
  g1[2] = (int)(((tensor_units >> 16) & 0xFFFFu) | (1u << 16)); // d0 hi | d1=1
  g1[3] = (int)((tile_units & 0xFFFFu) << 16);              // tile_dim0
  g1[4] = 0; g1[5] = 0; g1[6] = 0; g1[7] = 0;
  i32x4 z4; z4[0] = 0; z4[1] = 0; z4[2] = 0; z4[3] = 0;
  i32x8 z8;
#pragma unroll
  for (int i = 0; i < 8; ++i) z8[i] = 0;
  __builtin_amdgcn_tensor_load_to_lds(g0, g1, z4, z4, z8, 0);
}

__device__ __forceinline__ void tdm_load_2d(const void* gptr, void* lptr,
                                            unsigned tile0, unsigned tile1,
                                            unsigned stride0,
                                            unsigned td0, unsigned td1) {
  const unsigned long long ga = (unsigned long long)(uintptr_t)gptr;
  const unsigned la = (unsigned)(uintptr_t)lptr;
  u32x4 g0;
  g0[0] = 1u;
  g0[1] = la;
  g0[2] = (unsigned)ga;
  g0[3] = (unsigned)((ga >> 32) & 0x01FFFFFFu) | (2u << 30);
  i32x8 g1;
  g1[0] = (int)(3u << 16);                                  // data_size=8B
  g1[1] = (int)((td0 & 0xFFFFu) << 16);
  g1[2] = (int)(((td0 >> 16) & 0xFFFFu) | ((td1 & 0xFFFFu) << 16));
  g1[3] = (int)(((td1 >> 16) & 0xFFFFu) | ((tile0 & 0xFFFFu) << 16));
  g1[4] = (int)(tile1 & 0xFFFFu);                           // tile_dim1
  g1[5] = (int)stride0;                                     // dim0_stride lo32
  g1[6] = 0;                                                // stride hi bits
  g1[7] = 0;
  i32x4 z4; z4[0] = 0; z4[1] = 0; z4[2] = 0; z4[3] = 0;
  i32x8 z8;
#pragma unroll
  for (int i = 0; i < 8; ++i) z8[i] = 0;
  __builtin_amdgcn_tensor_load_to_lds(g0, g1, z4, z4, z8, 0);
}
#endif  // USE_TDM

// ---------------------------------------------------------------------------
// Fused flash attention, double-buffered K/V tiles fed by the Tensor Data
// Mover. 4 waves, 64 query rows per block. CVT = number of 16-channel V tiles.
//   Q,K: [B, L, 256] f16 (position-major)   V: [B, CVT*16, L] f16 (ch-major)
// ---------------------------------------------------------------------------
template <int CVT>
__global__ __launch_bounds__(128) void attn_kernel(const _Float16* __restrict__ Q,
                                                   const _Float16* __restrict__ K,
                                                   const _Float16* __restrict__ V,
                                                   float* __restrict__ out0,
                                                   float* __restrict__ out1,
                                                   int cReal) {
  __shared__ __align__(128) _Float16 sK[2][64 * C_];        // 2 x 32 KB
  __shared__ __align__(128) _Float16 sV[2][CVT * 16 * 64];  // 2 x (CVT*2) KB
  __shared__ __align__(64)  _Float16 sP[4 * 16 * 64];       // 8 KB P staging

  const int tid  = threadIdx.x;
  const int lane = tid & 31;
  const int wave = tid >> 5;        // 0..3 -> 16-row slice
  const int b    = blockIdx.y;
  const int i0   = blockIdx.x * 64; // target row base
  const int lm   = lane & 15;
  const int hi16 = lane >> 4;       // 0 -> rows r, 1 -> rows r+8
  const int h8   = hi16 * 8;

  const _Float16* Kb = K + (size_t)b * L_ * C_;
  const _Float16* Vb = V + (size_t)b * (CVT * 16) * L_;

  // ---- Q A-fragments in registers (16 rows x 256 channels) ----
  const int qrow = i0 + wave * 16 + lm;
  const _Float16* qp = Q + ((size_t)b * L_ + qrow) * C_;
  v16h qa[8];
#pragma unroll
  for (int kk = 0; kk < 8; ++kk) {
    v8h lo = *(const v8h*)(qp + kk * 32 + hi16 * 16);
    v8h hi = *(const v8h*)(qp + kk * 32 + hi16 * 16 + 8);
#pragma unroll
    for (int e = 0; e < 8; ++e) { qa[kk][e] = lo[e]; qa[kk][e + 8] = hi[e]; }
  }

  float mrow[8], lrow[8];
  v8f O[CVT];
#pragma unroll
  for (int r = 0; r < 8; ++r) { mrow[r] = -1e30f; lrow[r] = 0.f; }
#pragma unroll
  for (int t = 0; t < CVT; ++t)
#pragma unroll
    for (int r = 0; r < 8; ++r) O[t][r] = 0.f;

  _Float16* pw = sP + wave * (16 * 64);
  const int NCHUNK = L_ / 64;

#if USE_TDM
  // Prologue: DMA chunk 0 into buffer 0 (wave 0 issues; TDM ignores EXEC).
  if (wave == 0) {
    tdm_load_1d(Kb, &sK[0][0], (64 * C_ * 2) / 8, (unsigned)(L_ * C_ * 2 / 8));
    tdm_load_2d(Vb, &sV[0][0], (64 * 2) / 8, CVT * 16, (L_ * 2) / 8,
                (L_ * 2) / 8, CVT * 16);
  }
#endif

  for (int j = 0; j < NCHUNK; ++j) {
    const int cur = j & 1;
#if USE_TDM
    if (wave == 0) {
      if (j + 1 < NCHUNK) {
        // Prefetch chunk j+1 into the other buffer (its readers finished at
        // the trailing barrier of iteration j-1), then wait for chunk j's two
        // descriptors (allow the two just-issued ones to stay in flight).
        tdm_load_1d(Kb + (size_t)(j + 1) * 64 * C_, &sK[1 - cur][0],
                    (64 * C_ * 2) / 8, (unsigned)(L_ * C_ * 2 / 8));
        tdm_load_2d(Vb + (size_t)(j + 1) * 64, &sV[1 - cur][0],
                    (64 * 2) / 8, CVT * 16, (L_ * 2) / 8,
                    (L_ * 2) / 8, CVT * 16);
        __builtin_amdgcn_s_wait_tensorcnt((short)2);
      } else {
        __builtin_amdgcn_s_wait_tensorcnt((short)0);
      }
    }
    __syncthreads();  // chunk j resident for all waves
#else
    __syncthreads();
    {  // cooperative fallback loads
      const v8h* src = (const v8h*)(Kb + (size_t)j * 64 * C_);
      v8h* dst = (v8h*)&sK[cur][0];
#pragma unroll
      for (int i = 0; i < 16; ++i) dst[tid + i * 128] = src[tid + i * 128];
#pragma unroll
      for (int i = 0; i < CVT; ++i) {
        const int idx = tid + i * 128;
        const int row = idx >> 3, col = idx & 7;
        *(v8h*)&sV[cur][row * 64 + col * 8] =
            *(const v8h*)(Vb + (size_t)row * L_ + j * 64 + col * 8);
      }
    }
    __syncthreads();
#endif

    const _Float16* kb = &sK[cur][0];
    const _Float16* vb = &sV[cur][0];

    // ---- S = Q * K^T : kk outer / nt inner keeps 4 independent WMMA
    //      accumulation chains in flight (XDL pipelining) and reuses the
    //      qa[kk] A-operand across 4 consecutive WMMAs. ----
    v8f S[4];
#pragma unroll
    for (int nt = 0; nt < 4; ++nt)
#pragma unroll
      for (int r = 0; r < 8; ++r) S[nt][r] = 0.f;
#pragma unroll
    for (int kk = 0; kk < 8; ++kk) {
#pragma unroll
      for (int nt = 0; nt < 4; ++nt) {
        const int kpos = nt * 16 + lm;
        v16h bfrag = *(const v16h*)&kb[kpos * C_ + kk * 32 + hi16 * 16];
        S[nt] = __builtin_amdgcn_wmma_f32_16x16x32_f16(false, qa[kk], false, bfrag,
                                                       (short)0, S[nt], false, false);
      }
    }

    // ---- online softmax (row stats within each 16-lane group) ----
    float nm[8], alpha[8];
#pragma unroll
    for (int r = 0; r < 8; ++r) {
      float mx = fmaxf(fmaxf(S[0][r], S[1][r]), fmaxf(S[2][r], S[3][r]));
#pragma unroll
      for (int m = 1; m < 16; m <<= 1) mx = fmaxf(mx, __shfl_xor(mx, m, 32));
      nm[r]    = fmaxf(mrow[r], mx);
      alpha[r] = __expf(mrow[r] - nm[r]);
      mrow[r]  = nm[r];
    }
#pragma unroll
    for (int r = 0; r < 8; ++r) {
      float rs = 0.f;
#pragma unroll
      for (int nt = 0; nt < 4; ++nt) {
        float p = __expf(S[nt][r] - nm[r]);
        rs += p;
        pw[(r + h8) * 64 + nt * 16 + lm] = (_Float16)p;  // P -> LDS (f16)
      }
#pragma unroll
      for (int m = 1; m < 16; m <<= 1) rs += __shfl_xor(rs, m, 32);
      lrow[r] = lrow[r] * alpha[r] + rs;
    }
#pragma unroll
    for (int t = 0; t < CVT; ++t)
#pragma unroll
      for (int r = 0; r < 8; ++r) O[t][r] *= alpha[r];

    asm volatile("s_wait_dscnt 0" ::: "memory");  // P visible within the wave

    // ---- O += P (16x64) * V (64 x CVT*16) ----
#pragma unroll
    for (int ks = 0; ks < 2; ++ks) {
      v8h lo = *(const v8h*)(pw + lm * 64 + ks * 32 + hi16 * 16);
      v8h hi = *(const v8h*)(pw + lm * 64 + ks * 32 + hi16 * 16 + 8);
      v16h pa;
#pragma unroll
      for (int e = 0; e < 8; ++e) { pa[e] = lo[e]; pa[e + 8] = hi[e]; }
#pragma unroll
      for (int t = 0; t < CVT; ++t) {
        v16h bfrag = *(const v16h*)&vb[(t * 16 + lm) * 64 + ks * 32 + hi16 * 16];
        O[t] = __builtin_amdgcn_wmma_f32_16x16x32_f16(false, pa, false, bfrag,
                                                      (short)0, O[t], false, false);
      }
    }
    __syncthreads();  // compute done: next iteration may overwrite buffers
  }

  // ---- epilogue: normalize by softmax denom, scatter to outputs ----
#pragma unroll
  for (int t = 0; t < CVT; ++t) {
    const int ch = t * 16 + lm;
    if (ch >= cReal) continue;
    float* outp = (ch < 4) ? (out0 + ((size_t)b * 4 + ch) * L_)
                           : (out1 + ((size_t)b * 151 + (ch - 4)) * L_);
#pragma unroll
    for (int r = 0; r < 8; ++r) {
      const int row = i0 + wave * 16 + r + h8;
      outp[row] = O[t][r] / lrow[r];
    }
  }
}

// ---------------------------------------------------------------------------
extern "C" void kernel_launch(void* const* d_in, const int* in_sizes, int n_in,
                              void* d_out, int out_size, void* d_ws, size_t ws_size,
                              hipStream_t stream) {
  const float* trg  = (const float*)d_in[0];
  const float* src  = (const float*)d_in[1];
  const float* ref  = (const float*)d_in[2];
  const float* sf   = (const float*)d_in[3];
  const float* rf   = (const float*)d_in[4];
  const float* rsem = (const float*)d_in[5];
  float* out = (float*)d_out;

  // Workspace layout (f16): Qt | Ks | Kr | Vs(16ch) | Vr(160ch)  (~14.8 MB)
  _Float16* ws = (_Float16*)d_ws;
  const size_t nQ = (size_t)B_ * L_ * C_;
  _Float16* Qt = ws;
  _Float16* Ks = Qt + nQ;
  _Float16* Kr = Ks + nQ;
  _Float16* Vs = Kr + nQ;
  _Float16* Vr = Vs + (size_t)B_ * 16 * L_;

  const int normBlocks = (B_ * L_) / 8;  // 8 waves per 256-thread block
  norm_kernel<<<normBlocks, 256, 0, stream>>>(trg, Qt);
  norm_kernel<<<normBlocks, 256, 0, stream>>>(src, Ks);
  norm_kernel<<<normBlocks, 256, 0, stream>>>(ref, Kr);

  pack_src_v<<<(B_ * 16 * L_) / 256, 256, 0, stream>>>(sf, Vs);
  pack_ref_v<<<(B_ * 160 * L_) / 256, 256, 0, stream>>>(rf, rsem, Vr);

  dim3 grid(L_ / 64, B_);
  attn_kernel<1><<<grid, 128, 0, stream>>>(Qt, Ks, Vs,
                                           out, (float*)nullptr, 4);
  attn_kernel<10><<<grid, 128, 0, stream>>>(Qt, Kr, Vr,
                                            out + (size_t)B_ * 4 * L_,
                                            out + (size_t)2 * B_ * 4 * L_, 155);
}
